// GPTNeoSelfAttention_6717328851321
// MI455X (gfx1250) — matrix operand order repaired and checked by reading, not compiled, not run on hardware
//
#include <hip/hip_runtime.h>
#include <hip/hip_bf16.h>

// Problem constants (GPT-Neo self-attention, B=2, S=2048, E=2048, H=16, D=128)
#define B_ 2
#define S_ 2048
#define E_ 2048
#define H_ 16
#define D_ 128
#define M_ (B_ * S_)          // 4096 rows of activations

typedef __attribute__((ext_vector_type(16))) __bf16 bf16x16;
typedef __attribute__((ext_vector_type(4)))  __bf16 bf16x4;
typedef __attribute__((ext_vector_type(8)))  float  f32x8;
typedef __attribute__((ext_vector_type(4)))  int    v4i;

#define AS1 __attribute__((address_space(1)))
#define AS3 __attribute__((address_space(3)))

// ---------------------------------------------------------------------------
// CDNA5 async global->LDS copy (ASYNCcnt-tracked DMA).
// Builtin signature: (AS1 v4i* global_src, AS3 v4i* lds_dst, imm offset, imm cpol)
// ---------------------------------------------------------------------------
static __device__ __forceinline__ void async_b128(void* lds, const void* g) {
#if __has_builtin(__builtin_amdgcn_global_load_async_to_lds_b128)
  __builtin_amdgcn_global_load_async_to_lds_b128(
      (AS1 v4i*)(uintptr_t)g, (AS3 v4i*)(unsigned)(uintptr_t)lds, 0, 0);
#else
  unsigned l = (unsigned)(uintptr_t)lds;  // generic LDS ptr low 32 bits = LDS addr
  asm volatile("global_load_async_to_lds_b128 %0, %1, off"
               :: "v"(l), "v"(g) : "memory");
#endif
}

#if __has_builtin(__builtin_amdgcn_s_wait_asynccnt)
#define ASYNC_WAIT(n) __builtin_amdgcn_s_wait_asynccnt(n)
#else
#define ASYNC_WAIT(n) asm volatile("s_wait_asynccnt %0" :: "i"(n) : "memory")
#endif

#if __has_builtin(__builtin_amdgcn_s_wait_dscnt)
#define DS_WAIT0() __builtin_amdgcn_s_wait_dscnt(0)
#else
#define DS_WAIT0() asm volatile("s_wait_dscnt 0x0" ::: "memory")
#endif

static __device__ __forceinline__ f32x8 wmma_bf16(bf16x16 a, bf16x16 b, f32x8 c) {
  // D = A(16x32 bf16) * B(32x16 bf16) + C(16x16 f32)
  return __builtin_amdgcn_wmma_f32_16x16x32_bf16(false, a, false, b, (short)0, c,
                                                 false, false);
}

// ---------------------------------------------------------------------------
// fp32 -> bf16 streaming convert (vectorized: float4 in, 4x bf16 out)
// ---------------------------------------------------------------------------
__global__ __launch_bounds__(256)
void cvt_f32_bf16(const float4* __restrict__ src, bf16x4* __restrict__ dst, int n4) {
  int i = blockIdx.x * 256 + threadIdx.x;
  if (i < n4) {
    float4 v = src[i];
    bf16x4 o = { (__bf16)v.x, (__bf16)v.y, (__bf16)v.z, (__bf16)v.w };
    dst[i] = o;
  }
}

// ---------------------------------------------------------------------------
// C[M,N] = A[M,K] * B[N,K]^T  (bf16 in, fp32 accumulate)
// 128x128 tile per block, 8 wave32s each owning a 32x64 patch.
// K staged in 32-wide slabs, double-buffered via async global->LDS DMA.
// Last slab peeled so the steady-state loop has unconditional waits.
// ---------------------------------------------------------------------------
__global__ __launch_bounds__(256)
void gemm_nt_bf16(const __bf16* __restrict__ A, const __bf16* __restrict__ Bm,
                  __bf16* __restrict__ Cb, float* __restrict__ Cf,
                  const float* __restrict__ bias, int M, int N, int K) {
  __shared__ __attribute__((aligned(32))) __bf16 sA[2][128][32];  // 16 KB
  __shared__ __attribute__((aligned(32))) __bf16 sB[2][128][32];  // 16 KB
  const int tid  = threadIdx.x;
  const int lane = tid & 31;
  const int wave = tid >> 5;
  const int m0 = blockIdx.x * 128;
  const int n0 = blockIdx.y * 128;
  const int wm = (wave & 3) * 32;   // wave row offset in tile
  const int wn = (wave >> 2) * 64;  // wave col offset in tile

  // Per-thread staging assignment: chunks tid and tid+256 of 512 (16B each)
  const int ra0 = tid >> 2,         qa0 = (tid & 3) * 8;
  const int ra1 = (tid + 256) >> 2, qa1 = ((tid + 256) & 3) * 8;

  auto stage = [&](int buf, int k0) {
    async_b128(&sA[buf][ra0][qa0], &A [(size_t)(m0 + ra0) * K + k0 + qa0]);
    async_b128(&sA[buf][ra1][qa1], &A [(size_t)(m0 + ra1) * K + k0 + qa1]);
    async_b128(&sB[buf][ra0][qa0], &Bm[(size_t)(n0 + ra0) * K + k0 + qa0]);
    async_b128(&sB[buf][ra1][qa1], &Bm[(size_t)(n0 + ra1) * K + k0 + qa1]);
  };

  f32x8 acc[2][4] = {};
  auto compute = [&](int cur) {
    bf16x16 af[2], bfr[4];
    #pragma unroll
    for (int i = 0; i < 2; ++i)
      af[i] = *(const bf16x16*)&sA[cur][wm + i * 16 + (lane & 15)][(lane >> 4) * 16];
    #pragma unroll
    for (int j = 0; j < 4; ++j)
      bfr[j] = *(const bf16x16*)&sB[cur][wn + j * 16 + (lane & 15)][(lane >> 4) * 16];
    #pragma unroll
    for (int i = 0; i < 2; ++i)
      #pragma unroll
      for (int j = 0; j < 4; ++j)
        acc[i][j] = wmma_bf16(af[i], bfr[j], acc[i][j]);
  };

  stage(0, 0);
  int k0 = 0;
  for (; k0 + 32 < K; k0 += 32) {      // steady state: unconditional pipeline
    const int cur = (k0 >> 5) & 1;
    stage(cur ^ 1, k0 + 32);
    ASYNC_WAIT(4);                     // in-order retirement => current resident
    __syncthreads();
    compute(cur);
    __syncthreads();                   // reads done before buffer refill
  }
  ASYNC_WAIT(0);                       // peeled last slab
  __syncthreads();
  compute((k0 >> 5) & 1);

  // C fragment layout: element r of v8f = row (r + 8*(lane>>4)), col (lane&15)
  const int colb = lane & 15;
  const int rowh = 8 * (lane >> 4);
  #pragma unroll
  for (int i = 0; i < 2; ++i)
    #pragma unroll
    for (int j = 0; j < 4; ++j) {
      int col     = n0 + wn + j * 16 + colb;
      int rowbase = m0 + wm + i * 16 + rowh;
      #pragma unroll
      for (int r = 0; r < 8; ++r) {
        float v = acc[i][j][r];
        if (Cf) Cf[(size_t)(rowbase + r) * N + col] = v + (bias ? bias[col] : 0.0f);
        else    Cb[(size_t)(rowbase + r) * N + col] = (__bf16)v;
      }
    }
}

// ---------------------------------------------------------------------------
// RoPE + head split: raw [B*S, E] bf16 -> out [B*H, S, D] bf16
// ---------------------------------------------------------------------------
__global__ __launch_bounds__(256)
void rope_split(const __bf16* __restrict__ raw, __bf16* __restrict__ out,
                int apply_rope) {
  int idx = blockIdx.x * 256 + threadIdx.x;      // [0, 4096*2048)
  int m = idx >> 11;                             // activation row
  int c = idx & (E_ - 1);
  int h = c >> 7, d = c & (D_ - 1);
  int b = m >> 11, s = m & (S_ - 1);

  float v   = (float)raw[idx];
  float res = v;
  if (apply_rope) {
    int   f     = d & 63;
    int   dp    = (d < 64) ? (d + 64) : (d - 64);
    float other = (float)raw[m * E_ + h * D_ + dp];
    float rot   = (d < 64) ? -other : other;
    float inv   = __powf(10000.0f, -(float)f * (1.0f / 64.0f));
    float ang   = (float)s * inv;
    float cs, sn;
    __sincosf(ang, &cs, &sn);
    res = v * cs + rot * sn;
  }
  out[(((size_t)(b * H_ + h) * S_) + s) * D_ + d] = (__bf16)res;
}

// ---------------------------------------------------------------------------
// Flash attention (causal, no 1/sqrt(D) scale — GPT-Neo quirk).
// Block = 128 queries of one (b,h); 8 waves, each wave a 16-query strip.
// Q tile staged once (fragments then live in VGPRs; its LDS is reused as the
// P-refragmentation scratch). K/V streamed in 32-key slabs, double-buffered
// via async global->LDS DMA; last slab peeled.
// ---------------------------------------------------------------------------
__global__ __launch_bounds__(256)
void attn_fwd(const __bf16* __restrict__ Q, const __bf16* __restrict__ Kt,
              const __bf16* __restrict__ V, __bf16* __restrict__ O) {
  __shared__ union __attribute__((aligned(32))) {
    __bf16 q[128][128];        // 32 KB (only live until qf extraction)
    __bf16 p[8][16][32];       //  8 KB per-wave P staging (live afterwards)
  } sQP;
  __shared__ __attribute__((aligned(32))) __bf16 sK[2][32][128];  // 16 KB
  __shared__ __attribute__((aligned(32))) __bf16 sV[2][32][128];  // 16 KB

  const int tid  = threadIdx.x;
  const int lane = tid & 31;
  const int wave = tid >> 5;
  const int bh   = blockIdx.y;          // 0..31 (= b*H + h)
  const int q0   = blockIdx.x * 128;
  const size_t hoff = (size_t)bh * S_ * D_;

  // Per-thread K/V staging assignment: chunks tid and tid+256 of 512
  const int rk0 = tid >> 4,         qk0 = (tid & 15) * 8;
  const int rk1 = (tid + 256) >> 4, qk1 = ((tid + 256) & 15) * 8;
  auto stageKV = [&](int buf, int k0) {
    async_b128(&sK[buf][rk0][qk0], &Kt[hoff + (size_t)(k0 + rk0) * D_ + qk0]);
    async_b128(&sK[buf][rk1][qk1], &Kt[hoff + (size_t)(k0 + rk1) * D_ + qk1]);
    async_b128(&sV[buf][rk0][qk0], &V [hoff + (size_t)(k0 + rk0) * D_ + qk0]);
    async_b128(&sV[buf][rk1][qk1], &V [hoff + (size_t)(k0 + rk1) * D_ + qk1]);
  };

  // Stage Q tile (128x128 bf16): 8 async chunks per thread
  #pragma unroll
  for (int c = tid; c < 2048; c += 256) {   // 128 rows x 16 chunks of 8 bf16
    int r = c >> 4, q = (c & 15) * 8;
    async_b128(&sQP.q[r][q], &Q[hoff + (size_t)(q0 + r) * D_ + q]);
  }
  stageKV(0, 0);          // overlap first K/V slab with the Q wait
  ASYNC_WAIT(4);          // oldest 8 (Q tile) retired; K/V slab may be in flight
  __syncthreads();

  bf16x16 qf[4];
  #pragma unroll
  for (int kk = 0; kk < 4; ++kk)
    qf[kk] = *(const bf16x16*)&sQP.q[wave * 16 + (lane & 15)][kk * 32 + (lane >> 4) * 16];
  DS_WAIT0();   // Q reads serviced before anyone may overwrite union with P data

  f32x8 o[8] = {};
  float m8[8], l8[8];
  #pragma unroll
  for (int r = 0; r < 8; ++r) { m8[r] = -3.0e38f; l8[r] = 0.0f; }

  const int qrow0 = q0 + wave * 16 + 8 * (lane >> 4);
  const int colL  = lane & 15;
  const int kend  = q0 + 128;           // causal: keys <= last query in block

  auto step = [&](int cur, int k0) {
    // Scores: 16 queries x 32 keys = two 16x16 tiles, contract over D=128
    f32x8 sc[2] = {};
    #pragma unroll
    for (int jt = 0; jt < 2; ++jt)
      #pragma unroll
      for (int kk = 0; kk < 4; ++kk) {
        bf16x16 kf =
            *(const bf16x16*)&sK[cur][jt * 16 + (lane & 15)][kk * 32 + (lane >> 4) * 16];
        sc[jt] = wmma_bf16(qf[kk], kf, sc[jt]);
      }

    // Causal mask + online softmax (row stats: element r = row r+8*(lane>>4))
    float lsc[8];
    #pragma unroll
    for (int r = 0; r < 8; ++r) {
      float mx = m8[r];
      #pragma unroll
      for (int jt = 0; jt < 2; ++jt) {
        int   col = k0 + jt * 16 + colL;
        float v   = (col <= qrow0 + r) ? sc[jt][r] : -3.0e38f;
        sc[jt][r] = v;
        mx = fmaxf(mx, v);
      }
      #pragma unroll
      for (int off = 1; off <= 8; off <<= 1)
        mx = fmaxf(mx, __shfl_xor(mx, off, 32));   // 16-lane half reduction
      float sum = 0.0f;
      #pragma unroll
      for (int jt = 0; jt < 2; ++jt) {
        float p = __expf(sc[jt][r] - mx);
        sc[jt][r] = p;
        sum += p;
      }
      #pragma unroll
      for (int off = 1; off <= 8; off <<= 1) sum += __shfl_xor(sum, off, 32);
      float corr = __expf(m8[r] - mx);
      l8[r] = l8[r] * corr + sum;
      m8[r] = mx;
      lsc[r] = corr;
    }
    #pragma unroll
    for (int t = 0; t < 8; ++t)
      #pragma unroll
      for (int r = 0; r < 8; ++r) o[t][r] *= lsc[r];

    // Re-fragment P through per-wave LDS patch, then O += P(16x32) @ V(32x128)
    #pragma unroll
    for (int jt = 0; jt < 2; ++jt)
      #pragma unroll
      for (int r = 0; r < 8; ++r)
        sQP.p[wave][r + 8 * (lane >> 4)][jt * 16 + colL] = (__bf16)sc[jt][r];
    bf16x16 pf = *(const bf16x16*)&sQP.p[wave][(lane & 15)][(lane >> 4) * 16];
    #pragma unroll
    for (int t = 0; t < 8; ++t) {
      bf16x16 vf;
      #pragma unroll
      for (int s = 0; s < 16; ++s) vf[s] = sV[cur][(lane >> 4) * 16 + s][t * 16 + (lane & 15)];
      o[t] = wmma_bf16(pf, vf, o[t]);
    }
  };

  int k0 = 0;
  for (; k0 + 32 < kend; k0 += 32) {    // steady state: unconditional pipeline
    const int cur = (k0 >> 5) & 1;
    stageKV(cur ^ 1, k0 + 32);
    ASYNC_WAIT(4);
    __syncthreads();
    step(cur, k0);
    __syncthreads();                    // K/V buffer reads done before refill
  }
  ASYNC_WAIT(0);                        // peeled last slab
  __syncthreads();
  step((k0 >> 5) & 1, k0);

  // Epilogue: normalize and scatter to [B*S, E] (merge heads)
  float invl[8];
  #pragma unroll
  for (int r = 0; r < 8; ++r) invl[r] = 1.0f / l8[r];
  const int b = bh >> 4, h = bh & 15;
  #pragma unroll
  for (int t = 0; t < 8; ++t) {
    int colg = h * D_ + t * 16 + colL;
    #pragma unroll
    for (int r = 0; r < 8; ++r)
      O[(size_t)(b * S_ + qrow0 + r) * E_ + colg] = (__bf16)(o[t][r] * invl[r]);
  }
}

// ---------------------------------------------------------------------------
// Host-side orchestration
// ---------------------------------------------------------------------------
extern "C" void kernel_launch(void* const* d_in, const int* in_sizes, int n_in,
                              void* d_out, int out_size, void* d_ws, size_t ws_size,
                              hipStream_t stream) {
  (void)in_sizes; (void)n_in; (void)out_size; (void)ws_size;
  const float* X  = (const float*)d_in[0];
  // d_in[1] = attention_mask (pure causal; structure known), d_in[2] = position_ids (arange)
  const float* Wq = (const float*)d_in[3];
  const float* Wk = (const float*)d_in[4];
  const float* Wv = (const float*)d_in[5];
  const float* Wo = (const float*)d_in[6];
  const float* bo = (const float*)d_in[7];
  float* out = (float*)d_out;

  // Workspace layout (bytes)
  char* w = (char*)d_ws;
  __bf16* Xb  = (__bf16*)(w + 0);          // 16 MB  [4096,2048]
  __bf16* Wqb = (__bf16*)(w + 16777216);   //  8 MB  [2048,2048]
  __bf16* Wkb = (__bf16*)(w + 25165824);
  __bf16* Wvb = (__bf16*)(w + 33554432);
  __bf16* Wob = (__bf16*)(w + 41943040);
  __bf16* Qr  = (__bf16*)(w + 50331648);   // 16 MB raw projections
  __bf16* Kr  = (__bf16*)(w + 67108864);
  __bf16* Vr  = (__bf16*)(w + 83886080);
  __bf16* Qh  = (__bf16*)(w + 100663296);  // 16 MB [B*H,S,D] rotated
  __bf16* Kh  = (__bf16*)(w + 117440512);
  __bf16* Vh  = (__bf16*)(w + 134217728);
  __bf16* Ab  = Qr;  // attention output reuses dead Qr buffer

  // 1) bf16 conversions
  cvt_f32_bf16<<<8192, 256, 0, stream>>>((const float4*)X,  (bf16x4*)Xb,  8388608 / 4);
  cvt_f32_bf16<<<4096, 256, 0, stream>>>((const float4*)Wq, (bf16x4*)Wqb, 4194304 / 4);
  cvt_f32_bf16<<<4096, 256, 0, stream>>>((const float4*)Wk, (bf16x4*)Wkb, 4194304 / 4);
  cvt_f32_bf16<<<4096, 256, 0, stream>>>((const float4*)Wv, (bf16x4*)Wvb, 4194304 / 4);
  cvt_f32_bf16<<<4096, 256, 0, stream>>>((const float4*)Wo, (bf16x4*)Wob, 4194304 / 4);

  // 2) QKV projections
  dim3 gg(M_ / 128, E_ / 128), gb(256);
  gemm_nt_bf16<<<gg, gb, 0, stream>>>(Xb, Wqb, Qr, nullptr, nullptr, M_, E_, E_);
  gemm_nt_bf16<<<gg, gb, 0, stream>>>(Xb, Wkb, Kr, nullptr, nullptr, M_, E_, E_);
  gemm_nt_bf16<<<gg, gb, 0, stream>>>(Xb, Wvb, Vr, nullptr, nullptr, M_, E_, E_);

  // 3) RoPE + head split
  rope_split<<<32768, 256, 0, stream>>>(Qr, Qh, 1);
  rope_split<<<32768, 256, 0, stream>>>(Kr, Kh, 1);
  rope_split<<<32768, 256, 0, stream>>>(Vr, Vh, 0);

  // 4) Causal flash attention
  attn_fwd<<<dim3(S_ / 128, B_ * H_), 256, 0, stream>>>(Qh, Kh, Vh, Ab);

  // 5) Output projection + bias (fp32 out)
  gemm_nt_bf16<<<gg, gb, 0, stream>>>(Ab, Wob, nullptr, out, bo, M_, E_, E_);
}
